// AdaptBlock_34359739092
// MI455X (gfx1250) — matrix operand order repaired and checked by reading, not compiled, not run on hardware
//
#include <hip/hip_runtime.h>
#include <hip/hip_bf16.h>
#include <stdint.h>

#define HH 128
#define WW 128
#define HWSZ 16384
#define CIN 64
#define COUT 64
#define NIMG 8
#define NPIX (NIMG * HWSZ) /* 131072 */
#define SA_STRIDE 72       /* halfs; 144 B rows, 16B aligned */
#define SB_STRIDE 72

typedef __attribute__((ext_vector_type(16))) __bf16 v16bf;
typedef __attribute__((ext_vector_type(8)))  float  v8f;
typedef __attribute__((ext_vector_type(8)))  unsigned int v8u;

__device__ __forceinline__ float bf2f(unsigned int h) {
    union { unsigned int u; float f; } v; v.u = (h & 0xFFFFu) << 16; return v.f;
}
__device__ __forceinline__ unsigned short f2bf(float f) {
    union { float f; unsigned int u; } v; v.f = f;
    unsigned int u = v.u;
    unsigned int r = u + 0x7FFFu + ((u >> 16) & 1u);   // round-to-nearest-even
    return (unsigned short)(r >> 16);
}

// ---------------- kernel 1: x NCHW f32 -> NHWC bf16 ----------------
__global__ void k_to_nhwc_bf16(const float* __restrict__ x, unsigned short* __restrict__ xbf) {
    int idx = blockIdx.x * 256 + threadIdx.x;          // over NPIX*CIN
    int c   = idx & (CIN - 1);
    int pix = idx >> 6;
    int n   = pix >> 14;
    int rem = pix & (HWSZ - 1);
    xbf[idx] = f2bf(x[(((size_t)(n * CIN + c)) << 14) + rem]);
}

// ---------------- kernel 2: dcn_w [O][C][9] f32 -> bprep [tap][o][c] bf16 ----------------
__global__ void k_prep_b(const float* __restrict__ dcn_w, unsigned short* __restrict__ bprep) {
    int idx = blockIdx.x * 256 + threadIdx.x;          // over 9*64*64
    int k = idx >> 12;
    int r = idx & 4095;
    int o = r >> 6;
    int c = r & 63;
    bprep[idx] = f2bf(dcn_w[(o * CIN + c) * 9 + k]);
}

// ---------------- kernel 3: per-pixel offsets -> packed coords + bilinear weights ----------------
__global__ void k_offsets(const float* __restrict__ x,
                          const float* __restrict__ tmw, const float* __restrict__ tmb,
                          const float* __restrict__ trw, const float* __restrict__ trb,
                          unsigned int* __restrict__ offYX, float4* __restrict__ offW) {
    int gp  = blockIdx.x * 256 + threadIdx.x;          // over NPIX
    int n   = gp >> 14;
    int rem = gp & (HWSZ - 1);
    int h   = rem >> 7;
    int w   = rem & (WW - 1);

    float t0 = tmb[0], t1 = tmb[1], t2 = tmb[2], t3 = tmb[3];
    float u0 = trb[0], u1 = trb[1];
    const float* xn = x + (((size_t)n * CIN) << 14);
    for (int c = 0; c < CIN; ++c) {
        const float* xc = xn + ((size_t)c << 14);
        const float* wm = tmw + c * 9;
        const float* wr = trw + c * 9;
#pragma unroll
        for (int tap = 0; tap < 9; ++tap) {
            int dy = tap / 3 - 1, dx = tap % 3 - 1;
            int yy = h + dy, xx = w + dx;
            float v = 0.f;
            if (yy >= 0 && yy < HH && xx >= 0 && xx < WW) v = xc[yy * WW + xx];
            t0 = fmaf(v, wm[tap],        t0);
            t1 = fmaf(v, wm[576 + tap],  t1);
            t2 = fmaf(v, wm[1152 + tap], t2);
            t3 = fmaf(v, wm[1728 + tap], t3);
            u0 = fmaf(v, wr[tap],        u0);
            u1 = fmaf(v, wr[576 + tap],  u1);
        }
    }

#pragma unroll
    for (int tap = 0; tap < 9; ++tap) {
        float r0 = (float)(tap / 3 - 1);
        float r1 = (float)(tap % 3 - 1);
        float offy = t0 * r0 + t1 * r1 - r0 + u0;
        float offx = t2 * r0 + t3 * r1 - r1 + u1;
        float py = (float)h + r0 + offy;
        float px = (float)w + r1 + offx;
        float y0f = floorf(py), x0f = floorf(px);
        float wy = py - y0f, wx = px - x0f;
        int y0 = (int)y0f, x0 = (int)x0f;
        int y1 = y0 + 1, x1 = x0 + 1;
        float vy0 = (y0 >= 0 && y0 < HH) ? 1.f : 0.f;
        float vy1 = (y1 >= 0 && y1 < HH) ? 1.f : 0.f;
        float vx0 = (x0 >= 0 && x0 < WW) ? 1.f : 0.f;
        float vx1 = (x1 >= 0 && x1 < WW) ? 1.f : 0.f;
        float4 wv;
        wv.x = (1.f - wy) * (1.f - wx) * vy0 * vx0;
        wv.y = (1.f - wy) * wx         * vy0 * vx1;
        wv.z = wy * (1.f - wx)         * vy1 * vx0;
        wv.w = wy * wx                 * vy1 * vx1;
        int y0c = min(max(y0, 0), HH - 1), y1c = min(max(y1, 0), HH - 1);
        int x0c = min(max(x0, 0), WW - 1), x1c = min(max(x1, 0), WW - 1);
        unsigned int pk = (unsigned)y0c | ((unsigned)y1c << 8) |
                          ((unsigned)x0c << 16) | ((unsigned)x1c << 24);
        offYX[gp * 9 + tap] = pk;
        offW [gp * 9 + tap] = wv;
    }
}

// ---------------- kernel 4: fused sample + bf16 WMMA GEMM + BN + residual + ReLU ----------------
__global__ __launch_bounds__(256)
void k_main(const unsigned short* __restrict__ xbf,
            const unsigned short* __restrict__ bprep,
            const unsigned int*   __restrict__ offYX,
            const float4*         __restrict__ offW,
            const float*          __restrict__ xres,
            const float* __restrict__ gamma, const float* __restrict__ beta,
            const float* __restrict__ mean,  const float* __restrict__ var,
            float* __restrict__ out) {
    __shared__ __align__(16) unsigned short sA[128 * SA_STRIDE]; // [pixel][channel] bf16
    __shared__ __align__(16) unsigned short sB[64  * SB_STRIDE]; // [o][c] bf16 (B^T)

    const int tid    = threadIdx.x;
    const int wgBase = blockIdx.x * 128;
    const int lane   = tid & 31;
    const int wv     = tid >> 5;          // wave 0..7, each owns 16 pixels

    v8f acc[4] = {v8f{}, v8f{}, v8f{}, v8f{}};

    // sampling mapping: 2 threads per pixel, 32 channels each
    const int p    = tid >> 1;
    const int part = tid & 1;
    const int gp   = wgBase + p;
    const int n_s  = gp >> 14;
    const unsigned short* xb_n = xbf + ((((size_t)n_s) << 14) * CIN);

    for (int tap = 0; tap < 9; ++tap) {
        if (tap < 8) __builtin_prefetch(bprep + (tap + 1) * 4096, 0, 1);

        // ---- stage A: bilinear sample 128 px x 64 ch into LDS (bf16)
        unsigned int pk = offYX[gp * 9 + tap];
        float4 w4 = offW[gp * 9 + tap];
        int y0 = pk & 255, y1 = (pk >> 8) & 255;
        int x0 = (pk >> 16) & 255, x1 = (pk >> 24) & 255;
        const unsigned short* r00 = xb_n + ((y0 * WW + x0) * CIN) + part * 32;
        const unsigned short* r01 = xb_n + ((y0 * WW + x1) * CIN) + part * 32;
        const unsigned short* r10 = xb_n + ((y1 * WW + x0) * CIN) + part * 32;
        const unsigned short* r11 = xb_n + ((y1 * WW + x1) * CIN) + part * 32;
        unsigned short* dst = &sA[p * SA_STRIDE + part * 32];
#pragma unroll
        for (int cb = 0; cb < 4; ++cb) {
            uint4 a0 = *(const uint4*)(r00 + cb * 8);
            uint4 a1 = *(const uint4*)(r01 + cb * 8);
            uint4 a2 = *(const uint4*)(r10 + cb * 8);
            uint4 a3 = *(const uint4*)(r11 + cb * 8);
            unsigned int res[4];
#pragma unroll
            for (int q = 0; q < 4; ++q) {
                unsigned int u0 = ((const unsigned int*)&a0)[q];
                unsigned int u1 = ((const unsigned int*)&a1)[q];
                unsigned int u2 = ((const unsigned int*)&a2)[q];
                unsigned int u3 = ((const unsigned int*)&a3)[q];
                float lo = w4.x * bf2f(u0) + w4.y * bf2f(u1) + w4.z * bf2f(u2) + w4.w * bf2f(u3);
                float hi = w4.x * bf2f(u0 >> 16) + w4.y * bf2f(u1 >> 16) +
                           w4.z * bf2f(u2 >> 16) + w4.w * bf2f(u3 >> 16);
                res[q] = (unsigned int)f2bf(lo) | ((unsigned int)f2bf(hi) << 16);
            }
            *(uint4*)(dst + cb * 8) = *(uint4*)res;
        }

        // ---- stage B chunk [64 o][64 c] into LDS
        {
            const uint4* src = (const uint4*)(bprep + tap * 4096);
#pragma unroll
            for (int rr = 0; rr < 2; ++rr) {
                int j = tid + rr * 256;            // 0..511 uint4 jobs
                int o = j >> 3, q = j & 7;
                *(uint4*)(&sB[o * SB_STRIDE + q * 8]) = src[j];
            }
        }
        __syncthreads();

        // ---- WMMA: this wave's 16-pixel M-tile x all 64 O, K-chunk of 64 (2 ksteps)
        const int mrow = wv * 16 + (lane & 15);
        const unsigned int* rowA = (const unsigned int*)(&sA[mrow * SA_STRIDE]);
        const int kg = (lane < 16) ? 0 : 8;
#pragma unroll
        for (int ks = 0; ks < 2; ++ks) {
            v8u au;
#pragma unroll
            for (int i = 0; i < 8; ++i) {
                // 16-bit A 16x32 layout: VGPR i holds K pair; lanes 16-31 shifted by 8
                int kp = (i < 4) ? (kg + 2 * i) : (16 + kg + 2 * (i - 4));
                au[i] = rowA[(ks * 32 + kp) >> 1];
            }
            v16bf afrag = __builtin_bit_cast(v16bf, au);
            const int kb = ks * 32 + ((lane < 16) ? 0 : 8);
#pragma unroll
            for (int t = 0; t < 4; ++t) {
                const uint4* rb = (const uint4*)(&sB[(t * 16 + (lane & 15)) * SB_STRIDE + kb]);
                uint4 b0 = rb[0], b1 = rb[2];
                v8u bu;
                bu[0] = b0.x; bu[1] = b0.y; bu[2] = b0.z; bu[3] = b0.w;
                bu[4] = b1.x; bu[5] = b1.y; bu[6] = b1.z; bu[7] = b1.w;
                v16bf bfrag = __builtin_bit_cast(v16bf, bu);
                acc[t] = __builtin_amdgcn_wmma_f32_16x16x32_bf16(
                    false, afrag, false, bfrag, (short)0, acc[t], false, false);
            }
        }
        __syncthreads();
    }

    // ---- epilogue: BN + residual + ReLU, scattered f32 stores
#pragma unroll
    for (int t = 0; t < 4; ++t) {
        int o = t * 16 + (lane & 15);
        float sc = gamma[o] * rsqrtf(var[o] + 1e-5f);
        float sh = beta[o] - mean[o] * sc;
#pragma unroll
        for (int r = 0; r < 8; ++r) {
            int m   = r + ((lane < 16) ? 0 : 8);       // C/D 32-bit layout
            int gpx = wgBase + wv * 16 + m;
            int n   = gpx >> 14;
            int rem = gpx & (HWSZ - 1);
            size_t idx = (((size_t)(n * COUT + o)) << 14) + rem;
            float v = acc[t][r] * sc + sh + xres[idx];
            out[idx] = v > 0.f ? v : 0.f;
        }
    }
}

extern "C" void kernel_launch(void* const* d_in, const int* in_sizes, int n_in,
                              void* d_out, int out_size, void* d_ws, size_t ws_size,
                              hipStream_t stream) {
    const float* x     = (const float*)d_in[0];
    const float* tm_w  = (const float*)d_in[1];
    const float* tm_b  = (const float*)d_in[2];
    const float* tr_w  = (const float*)d_in[3];
    const float* tr_b  = (const float*)d_in[4];
    const float* dcn_w = (const float*)d_in[5];
    const float* gam   = (const float*)d_in[6];
    const float* bet   = (const float*)d_in[7];
    const float* mea   = (const float*)d_in[8];
    const float* var   = (const float*)d_in[9];
    float* out = (float*)d_out;

    char* ws = (char*)d_ws;
    unsigned short* xbf   = (unsigned short*)ws;                                   // 16 MB
    unsigned short* bprep = (unsigned short*)(ws + 16777216);                      // 73728 B
    unsigned int*   offYX = (unsigned int*)  (ws + 16777216 + 73728);              // 4.5 MB
    float4*         offW  = (float4*)        (ws + 16777216 + 73728 + 4718592);    // 18 MB

    k_to_nhwc_bf16<<<(NPIX * CIN) / 256, 256, 0, stream>>>(x, xbf);
    k_prep_b<<<(9 * 64 * 64) / 256, 256, 0, stream>>>(dcn_w, bprep);
    k_offsets<<<NPIX / 256, 256, 0, stream>>>(x, tm_w, tm_b, tr_w, tr_b, offYX, offW);
    k_main<<<NPIX / 128, 256, 0, stream>>>(xbf, bprep, offYX, offW, x,
                                           gam, bet, mea, var, out);
}
